// Observation_51754355917478
// MI455X (gfx1250) — compile-verified, matrix operand order checked by reading, not run on hardware
//
#include <hip/hip_runtime.h>
#include <hip/hip_bf16.h>

typedef __bf16 bf16;
typedef __attribute__((ext_vector_type(8)))  __bf16 bf16x8;
typedef __attribute__((ext_vector_type(16))) __bf16 bf16x16;
typedef __attribute__((ext_vector_type(8)))  float  f32x8;
typedef __attribute__((ext_vector_type(2)))  float  f32x2;

#define BS     128
#define T_V    256
#define T_S    40
#define V_INK  1024
#define S_INK  300
#define S_PAD  320
#define VH     512
#define SH     256
#define LCROSS 512
#define NFEAT  1024
#define NCAT   2624   // 512 + 1024 + 512 + 512 + 64

static __device__ __forceinline__ float sigm(float x) { return 1.0f / (1.0f + __expf(-x)); }

static __device__ __forceinline__ bf16x16 mk16(const bf16* p0, const bf16* p1) {
  bf16x8 lo = *(const bf16x8*)p0;
  bf16x8 hi = *(const bf16x8*)p1;
  bf16x16 r;
#pragma unroll
  for (int i = 0; i < 8; ++i) { r[i] = lo[i]; r[i + 8] = hi[i]; }
  return r;
}

static __device__ __forceinline__ f32x8 cvt8(const bf16* p) {
  bf16x8 v = *(const bf16x8*)p;
  f32x8 r;
#pragma unroll
  for (int i = 0; i < 8; ++i) r[i] = (float)v[i];
  return r;
}

static __device__ __forceinline__ f32x8 wmma_bf16(bf16x16 a, bf16x16 b, f32x8 c) {
  return __builtin_amdgcn_wmma_f32_16x16x32_bf16(false, a, false, b, (short)0, c, false, false);
}

static __device__ __forceinline__ f32x8 wmma_f32(f32x2 a, f32x2 b, f32x8 c) {
  return __builtin_amdgcn_wmma_f32_16x16x4_f32(false, a, false, b, (short)0, c, false, false);
}

// ---------------- weight conversion ----------------
__global__ void k_cvt_bf16(const float* __restrict__ s, bf16* __restrict__ d, size_t n) {
  for (size_t i = (size_t)blockIdx.x * blockDim.x + threadIdx.x; i < n;
       i += (size_t)gridDim.x * blockDim.x)
    d[i] = (bf16)s[i];
}

__global__ void k_cvt_bf16_pad(const float* __restrict__ s, bf16* __restrict__ d,
                               int rows, int kin, int kout) {
  size_t n = (size_t)rows * kout;
  for (size_t i = (size_t)blockIdx.x * blockDim.x + threadIdx.x; i < n;
       i += (size_t)gridDim.x * blockDim.x) {
    int r = (int)(i / kout), k = (int)(i % kout);
    d[i] = (k < kin) ? (bf16)s[(size_t)r * kin + k] : (bf16)0.0f;
  }
}

// ---------------- input-projection GEMM (fully parallel over (t, bg, dir)) ----
// xp[d,b,t,:] = x[b,t,:] @ w_ih[d]^T + b_ih[d]   stored as bf16 C-fragments.
// grid: (TT, 16 = dir*8 + bg); block: HID threads (HID/32 waves).
// Fragment record: ((((d*8+bg)*TT + t)*NW + wave)*6 + g*2+t2)*256 + lane*8
template <int HID, int KIN, int KINR, int TT>
__global__ __launch_bounds__(HID) void k_xp_gemm(
    const float* __restrict__ x,    // [BS, TT, KINR] f32
    const bf16* __restrict__ w_ih,  // [2, 3H, KIN] bf16 (K padded)
    const float* __restrict__ b_ih, // [2, 3H]
    bf16* __restrict__ xpf)         // fragment buffer
{
  constexpr int NW = HID / 32;
  constexpr int NT = HID;
  constexpr int G3 = 3 * HID;
  __shared__ bf16 xs[16 * KIN];

  const int tid  = threadIdx.x;
  const int lane = tid & 31;
  const int wave = tid >> 5;
  const int half = lane >> 4;
  const int lr   = lane & 15;
  const int t    = blockIdx.x;
  const int dd   = blockIdx.y >> 3;
  const int bg   = blockIdx.y & 7;
  const int b0   = bg * 16;
  const int j0   = wave * 32;

  // stage x tile [16, KIN] as bf16 (zero-pad K)
  for (int i = tid; i < 16 * KIN; i += NT) {
    int row = i / KIN, k = i % KIN;
    float v = (k < KINR) ? x[((size_t)(b0 + row) * TT + t) * KINR + k] : 0.0f;
    xs[i] = (bf16)v;
  }
  __syncthreads();

  f32x8 acc[3][2];
#pragma unroll
  for (int g = 0; g < 3; ++g)
#pragma unroll
    for (int t2 = 0; t2 < 2; ++t2) {
      float bb = b_ih[dd * G3 + g * HID + j0 + t2 * 16 + lr];
#pragma unroll
      for (int i = 0; i < 8; ++i) acc[g][t2][i] = bb;
    }

  for (int k0 = 0; k0 < KIN; k0 += 32) {
    const bf16* ap = xs + lr * KIN + k0 + 8 * half;
    bf16x16 a = mk16(ap, ap + 16);
#pragma unroll
    for (int g = 0; g < 3; ++g)
#pragma unroll
      for (int t2 = 0; t2 < 2; ++t2) {
        const bf16* wp =
            w_ih + ((size_t)dd * G3 + g * HID + j0 + t2 * 16 + lr) * KIN + k0 + 8 * half;
        acc[g][t2] = wmma_bf16(a, mk16(wp, wp + 16), acc[g][t2]);
      }
  }

  size_t base = ((((size_t)(dd * 8 + bg) * TT + t) * NW + wave) * 6) * 256 + (size_t)lane * 8;
#pragma unroll
  for (int g = 0; g < 3; ++g)
#pragma unroll
    for (int t2 = 0; t2 < 2; ++t2) {
      bf16x8 o;
#pragma unroll
      for (int i = 0; i < 8; ++i) o[i] = (bf16)acc[g][t2][i];
      *(bf16x8*)(xpf + base + (g * 2 + t2) * 256) = o;
    }
}

// ---------------- recurrent scan (hp = h @ w_hh^T only) ----------------
// grid: (BS/16, 2 dirs); block: HID threads. Wave w owns hidden cols [w*32, w*32+32).
template <int HID, int TT>
__global__ __launch_bounds__(HID) void k_gru_scan(
    const bf16* __restrict__ xpf,   // precomputed fragments (b_ih folded in)
    const bf16* __restrict__ w_hh,  // [2, 3H, HID] bf16
    const float* __restrict__ b_hh, // [2, 3H]
    const int* __restrict__ slen,   // [BS]
    float* __restrict__ out)        // [BS, TT, 2H]; writes [b][t][d*H + c], masked
{
  constexpr int NW = HID / 32;
  constexpr int NT = HID;
  constexpr int G3 = 3 * HID;
  constexpr int FSTEP = NW * 6 * 256;  // fragment elements per (t) per (d,bg)
  __shared__ bf16 hs[16 * HID];
  __shared__ int  vls[16];

  const int tid  = threadIdx.x;
  const int lane = tid & 31;
  const int wave = tid >> 5;
  const int half = lane >> 4;
  const int lr   = lane & 15;
  const int bg   = blockIdx.x;
  const int b0   = bg * 16;
  const int dd   = blockIdx.y;
  const int j0   = wave * 32;

  if (tid < 16) vls[tid] = slen[b0 + tid];
  for (int i = tid; i < 16 * HID; i += NT) hs[i] = (bf16)0.0f;

  float hreg[2][8];
#pragma unroll
  for (int t2 = 0; t2 < 2; ++t2)
#pragma unroll
    for (int i = 0; i < 8; ++i) hreg[t2][i] = 0.0f;

  float brh[2], bzh[2], bnh[2];
#pragma unroll
  for (int t2 = 0; t2 < 2; ++t2) {
    int c = j0 + t2 * 16 + lr;
    brh[t2] = b_hh[dd * G3 + 0 * HID + c];
    bzh[t2] = b_hh[dd * G3 + 1 * HID + c];
    bnh[t2] = b_hh[dd * G3 + 2 * HID + c];
  }

  // wave-local fragment stream; backward dir walks te = TT-1..0 => stride -FSTEP
  const int te0 = dd ? (TT - 1) : 0;
  const ptrdiff_t fstride = dd ? -(ptrdiff_t)FSTEP : (ptrdiff_t)FSTEP;
  const bf16* fb = xpf + ((((size_t)(dd * 8 + bg) * TT + te0) * NW + wave) * 6) * 256 +
                   (size_t)lane * 8;

  __syncthreads();

  for (int t = 0; t < TT; ++t) {
    const int te = dd ? (TT - 1 - t) : t;

    // init accumulators from precomputed xp fragments
    f32x8 acc_r[2], acc_z[2], acc_xn[2], acc_hn[2];
#pragma unroll
    for (int t2 = 0; t2 < 2; ++t2) {
      acc_r[t2]  = cvt8(fb + (0 * 2 + t2) * 256);
      acc_z[t2]  = cvt8(fb + (1 * 2 + t2) * 256);
      acc_xn[t2] = cvt8(fb + (2 * 2 + t2) * 256);
#pragma unroll
      for (int i = 0; i < 8; ++i) {
        acc_r[t2][i] += brh[t2];
        acc_z[t2][i] += bzh[t2];
        acc_hn[t2][i] = bnh[t2];
      }
    }

    // prefetch next step's fragments (hides L2 latency across the barrier)
    if (t + 1 < TT) {
      const bf16* nf = fb + fstride;
#pragma unroll
      for (int f = 0; f < 6; ++f) __builtin_prefetch(nf + f * 256, 0, 3);
    }

    // hp = h @ w_hh^T   (K = HID)
    for (int k0 = 0; k0 < HID; k0 += 32) {
      const bf16* ap = hs + lr * HID + k0 + 8 * half;
      bf16x16 a = mk16(ap, ap + 16);
#pragma unroll
      for (int t2 = 0; t2 < 2; ++t2) {
        const int wr = j0 + t2 * 16 + lr;
        const bf16* w0 = w_hh + ((size_t)dd * G3 + 0 * HID + wr) * HID + k0 + 8 * half;
        const bf16* w1 = w_hh + ((size_t)dd * G3 + 1 * HID + wr) * HID + k0 + 8 * half;
        const bf16* w2 = w_hh + ((size_t)dd * G3 + 2 * HID + wr) * HID + k0 + 8 * half;
        acc_r[t2]  = wmma_bf16(a, mk16(w0, w0 + 16), acc_r[t2]);
        acc_z[t2]  = wmma_bf16(a, mk16(w1, w1 + 16), acc_z[t2]);
        acc_hn[t2] = wmma_bf16(a, mk16(w2, w2 + 16), acc_hn[t2]);
      }
    }
    __syncthreads();  // all hp reads of hs done before overwrite

    // gates + state update; C/D element i <-> (M = i + 8*half, N = lr)
#pragma unroll
    for (int t2 = 0; t2 < 2; ++t2) {
      const int c = j0 + t2 * 16 + lr;
#pragma unroll
      for (int i = 0; i < 8; ++i) {
        const int M = i + 8 * half;
        float r = sigm(acc_r[t2][i]);
        float z = sigm(acc_z[t2][i]);
        float n = tanhf(acc_xn[t2][i] + r * acc_hn[t2][i]);
        float hold = hreg[t2][i];
        float hnew = (1.0f - z) * n + z * hold;
        bool  mv   = te < vls[M];
        float hupd = mv ? hnew : hold;
        hreg[t2][i] = hupd;
        hs[M * HID + c] = (bf16)hupd;
        out[((size_t)(b0 + M) * TT + te) * (2 * HID) + dd * HID + c] = mv ? hnew : 0.0f;
      }
    }
    fb += fstride;
    __syncthreads();  // hs stable for next step
  }
}

// ---------------- sen_fea gather: words[b, lens[b]-1, :] ----------------
__global__ void k_sen_gather(const float* __restrict__ words, const int* __restrict__ lens,
                             float* __restrict__ senf) {
  int b = blockIdx.x;
  int c = blockIdx.y * 256 + threadIdx.x;  // 512 cols
  int t = lens[b] - 1;
  senf[(size_t)b * 512 + c] = words[((size_t)b * T_S + t) * 512 + c];
}

// ---------------- segment means + global mean over visual_fea ----------------
__global__ void k_segmean(const float* __restrict__ vf, const float* __restrict__ loc,
                          const int* __restrict__ vlen, float* __restrict__ leftv,
                          float* __restrict__ rightv, float* __restrict__ gvf) {
  int b = blockIdx.x, tid = threadIdx.x;  // 256 threads, 4 cols each
  float l0 = loc[b * 2 + 0], l1 = loc[b * 2 + 1];
  int   vl = vlen[b];
  float scale = (float)(vl - 1);
  bool  cond = (l0 <= l1);
  float le_f = cond ? l0 : 0.0f;
  float rs_f = cond ? l1 : 1.0f;
  int sl = 0, el = (int)floorf(le_f * scale);
  int sr = (int)floorf(rs_f * scale), er = (int)floorf(scale);
  bool okl = (sl <= el), okr = (sr <= er);
  float suml[4] = {0, 0, 0, 0}, sumr[4] = {0, 0, 0, 0}, sumg[4] = {0, 0, 0, 0};
  for (int t = 0; t < T_V; ++t) {
    float ml = (okl && t >= sl && t <= el) ? 1.0f : 0.0f;
    float mr = (okr && t >= sr && t <= er) ? 1.0f : 0.0f;
    float mg = (t < vl) ? 1.0f : 0.0f;
    const float* p = vf + ((size_t)b * T_V + t) * 1024;
#pragma unroll
    for (int u = 0; u < 4; ++u) {
      float v = p[tid + u * 256];
      suml[u] += ml * v; sumr[u] += mr * v; sumg[u] += mg * v;
    }
  }
  float cl = fmaxf(okl ? (float)(el - sl + 1) : 0.0f, 1.0f);
  float cr = fmaxf(okr ? (float)(er - sr + 1) : 0.0f, 1.0f);
  float cg = (float)vl;
#pragma unroll
  for (int u = 0; u < 4; ++u) {
    leftv [(size_t)b * 1024 + tid + u * 256] = suml[u] / cl;
    rightv[(size_t)b * 1024 + tid + u * 256] = sumr[u] / cr;
    gvf   [(size_t)b * 1024 + tid + u * 256] = sumg[u] / cg;
  }
}

// ---------------- concat fills ----------------
__global__ void k_cat_copy(const float* __restrict__ senf, const float* __restrict__ gvf,
                           float* __restrict__ cat) {
  int idx = blockIdx.x * 256 + threadIdx.x;  // 128*1536
  if (idx >= BS * 1536) return;
  int b = idx / 1536, c = idx % 1536;
  cat[(size_t)b * NCAT + c] = (c < 512) ? senf[(size_t)b * 512 + c]
                                        : gvf[(size_t)b * 1024 + (c - 512)];
}

__global__ void k_locfea(const float* __restrict__ loc, const float* __restrict__ w3,
                         const float* __restrict__ b3, float* __restrict__ cat) {
  int b = blockIdx.x, c = threadIdx.x;  // 64
  float v = loc[b * 2] * w3[c * 2] + loc[b * 2 + 1] * w3[c * 2 + 1] + b3[c];
  cat[(size_t)b * NCAT + 2560 + c] = v > 0.0f ? v : 0.0f;
}

// ---------------- head: fc1/fc2 fused (f32 WMMA 16x16x4, exact) ----------------
__global__ __launch_bounds__(32) void k_head_fc12(
    const float* __restrict__ senf, const float* __restrict__ leftv,
    const float* __restrict__ rightv, const float* __restrict__ fc1w,
    const float* __restrict__ fc1b, const float* __restrict__ fc2w,
    const float* __restrict__ fc2b, float* __restrict__ cat) {
  const int lane = threadIdx.x, half = lane >> 4, lr = lane & 15;
  const int r0 = blockIdx.x * 16, c0 = blockIdx.y * 16;
  f32x8 accP, accL, accR;
  float b2 = fc2b[c0 + lr], b1 = fc1b[c0 + lr];
#pragma unroll
  for (int i = 0; i < 8; ++i) { accP[i] = b2; accL[i] = b1; accR[i] = b1; }
  for (int k = 0; k < 512; k += 4) {
    f32x2 a = *(const f32x2*)(senf + (size_t)(r0 + lr) * 512 + k + 2 * half);
    f32x2 b = *(const f32x2*)(fc2w + (size_t)(c0 + lr) * 512 + k + 2 * half);
    accP = wmma_f32(a, b, accP);
  }
  for (int k = 0; k < 1024; k += 4) {
    f32x2 b  = *(const f32x2*)(fc1w + (size_t)(c0 + lr) * 1024 + k + 2 * half);
    f32x2 a  = *(const f32x2*)(leftv + (size_t)(r0 + lr) * 1024 + k + 2 * half);
    f32x2 a2 = *(const f32x2*)(rightv + (size_t)(r0 + lr) * 1024 + k + 2 * half);
    accL = wmma_f32(a, b, accL);
    accR = wmma_f32(a2, b, accR);
  }
#pragma unroll
  for (int i = 0; i < 8; ++i) {
    int M = i + 8 * half;
    float p = accP[i];
    float lv = accL[i] * p, rv = accR[i] * p;
    size_t row = (size_t)(r0 + M) * NCAT;
    cat[row + 1536 + c0 + lr] = lv > 0.0f ? lv : 0.0f;
    cat[row + 2048 + c0 + lr] = rv > 0.0f ? rv : 0.0f;
  }
}

// ---------------- head: fc4 (f32 WMMA 16x16x4) ----------------
__global__ __launch_bounds__(32) void k_head_fc4(const float* __restrict__ cat,
                                                 const float* __restrict__ fc4w,
                                                 const float* __restrict__ fc4b,
                                                 float* __restrict__ feat) {
  const int lane = threadIdx.x, half = lane >> 4, lr = lane & 15;
  const int r0 = blockIdx.x * 16, c0 = blockIdx.y * 16;
  f32x8 acc;
  float bb = fc4b[c0 + lr];
#pragma unroll
  for (int i = 0; i < 8; ++i) acc[i] = bb;
  for (int k = 0; k < NCAT; k += 4) {
    f32x2 a = *(const f32x2*)(cat + (size_t)(r0 + lr) * NCAT + k + 2 * half);
    f32x2 b = *(const f32x2*)(fc4w + (size_t)(c0 + lr) * NCAT + k + 2 * half);
    acc = wmma_f32(a, b, acc);
  }
#pragma unroll
  for (int i = 0; i < 8; ++i) {
    int M = i + 8 * half;
    float v = acc[i];
    feat[(size_t)(r0 + M) * NFEAT + c0 + lr] = v > 0.0f ? v : 0.0f;
  }
}

// ---------------- launch ----------------
extern "C" void kernel_launch(void* const* d_in, const int* in_sizes, int n_in,
                              void* d_out, int out_size, void* d_ws, size_t ws_size,
                              hipStream_t stream) {
  const float* gv     = (const float*)d_in[1];
  const float* sen    = (const float*)d_in[2];
  const float* loc    = (const float*)d_in[5];
  const int*   lens   = (const int*)d_in[6];
  const int*   vlen   = (const int*)d_in[7];
  const float* v_w_ih = (const float*)d_in[8];
  const float* v_w_hh = (const float*)d_in[9];
  const float* v_b_ih = (const float*)d_in[10];
  const float* v_b_hh = (const float*)d_in[11];
  const float* s_w_ih = (const float*)d_in[12];
  const float* s_w_hh = (const float*)d_in[13];
  const float* s_b_ih = (const float*)d_in[14];
  const float* s_b_hh = (const float*)d_in[15];
  const float* fc1w = (const float*)d_in[16];
  const float* fc1b = (const float*)d_in[17];
  const float* fc2w = (const float*)d_in[18];
  const float* fc2b = (const float*)d_in[19];
  const float* fc3w = (const float*)d_in[20];
  const float* fc3b = (const float*)d_in[21];
  const float* fc4w = (const float*)d_in[22];
  const float* fc4b = (const float*)d_in[23];

  float* out_vis  = (float*)d_out;                                 // [128,256,1024]
  float* out_senf = out_vis + (size_t)BS * T_V * 1024;             // [128,512]
  float* out_feat = out_senf + (size_t)BS * 512;                   // [128,1024]

  // workspace carve-up (bytes)
  char* w = (char*)d_ws;
  size_t o = 0;
  bf16* vwi = (bf16*)(w + o); o += (size_t)2 * 3 * VH * V_INK * 2;
  bf16* vwh = (bf16*)(w + o); o += (size_t)2 * 3 * VH * VH * 2;
  bf16* swi = (bf16*)(w + o); o += (size_t)2 * 3 * SH * S_PAD * 2;
  bf16* swh = (bf16*)(w + o); o += (size_t)2 * 3 * SH * SH * 2;
  bf16* xpv = (bf16*)(w + o); o += (size_t)2 * BS * T_V * 3 * VH * 2;   // 201 MB, frag layout
  bf16* xps = (bf16*)(w + o); o += (size_t)2 * BS * T_S * 3 * SH * 2;   // 15.7 MB
  float* words  = (float*)(w + o); o += (size_t)BS * T_S * 512 * 4;
  float* leftv  = (float*)(w + o); o += (size_t)BS * 1024 * 4;
  float* rightv = (float*)(w + o); o += (size_t)BS * 1024 * 4;
  float* gvf    = (float*)(w + o); o += (size_t)BS * 1024 * 4;
  float* cat    = (float*)(w + o); o += (size_t)BS * NCAT * 4;
  (void)o; (void)ws_size; (void)in_sizes; (void)n_in; (void)out_size;

  // weights -> bf16 (s_w_ih K padded 300 -> 320 with zeros)
  k_cvt_bf16<<<2048, 256, 0, stream>>>(v_w_ih, vwi, (size_t)2 * 3 * VH * V_INK);
  k_cvt_bf16<<<1024, 256, 0, stream>>>(v_w_hh, vwh, (size_t)2 * 3 * VH * VH);
  k_cvt_bf16<<<512,  256, 0, stream>>>(s_w_hh, swh, (size_t)2 * 3 * SH * SH);
  k_cvt_bf16_pad<<<512, 256, 0, stream>>>(s_w_ih, swi, 2 * 3 * SH, S_INK, S_PAD);

  // parallel input projections (full-chip WMMA GEMMs)
  k_xp_gemm<VH, V_INK, V_INK, T_V><<<dim3(T_V, 16), VH, 0, stream>>>(gv, vwi, v_b_ih, xpv);
  k_xp_gemm<SH, S_PAD, S_INK, T_S><<<dim3(T_S, 16), SH, 0, stream>>>(sen, swi, s_b_ih, xps);

  // recurrent scans (hp GEMM only on the sequential path)
  k_gru_scan<VH, T_V><<<dim3(BS / 16, 2), VH, 0, stream>>>(xpv, vwh, v_b_hh, vlen, out_vis);
  k_gru_scan<SH, T_S><<<dim3(BS / 16, 2), SH, 0, stream>>>(xps, swh, s_b_hh, lens, words);

  // sen_fea = words[b, lens-1]
  k_sen_gather<<<dim3(BS, 2), 256, 0, stream>>>(words, lens, out_senf);

  // segment/global means over visual_fea
  k_segmean<<<BS, 256, 0, stream>>>(out_vis, loc, vlen, leftv, rightv, gvf);

  // concat + head
  k_cat_copy<<<(BS * 1536 + 255) / 256, 256, 0, stream>>>(out_senf, gvf, cat);
  k_locfea<<<BS, 64, 0, stream>>>(loc, fc3w, fc3b, cat);
  k_head_fc12<<<dim3(BS / 16, LCROSS / 16), 32, 0, stream>>>(
      out_senf, leftv, rightv, fc1w, fc1b, fc2w, fc2b, cat);
  k_head_fc4<<<dim3(BS / 16, NFEAT / 16), 32, 0, stream>>>(cat, fc4w, fc4b, out_feat);
}
